// SARDecoder_66073776882245
// MI455X (gfx1250) — matrix operand order repaired and checked by reading, not compiled, not run on hardware
//
#include <hip/hip_runtime.h>
#include <hip/hip_bf16.h>
#include <stdint.h>
#include <stddef.h>

// ---------------- problem constants ----------------
#define BQ 128
#define CQ 512
#define HQ 8
#define WQ 32
#define RQ 512
#define AQ 512
#define V1Q 111
#define TQ 30
#define HWQ 256              // H*W
#define MPIX (BQ * HWQ)      // 32768 output pixels
#define NGATE 2048           // 4*R

typedef __attribute__((ext_vector_type(16))) __bf16 v16bf;
typedef __attribute__((ext_vector_type(8)))  float  v8f;

// ---------------- WMMA fragment helpers ----------------
// A-matrix 16x32 bf16 (ISA 7.12.2): lane L, g=L>>4, m=L&15 holds
//   elements 0..7  -> K = 8g + (0..7)        (contiguous)
//   elements 8..15 -> K = 16 + 8g + (0..7)   (contiguous, +16 elems from base)
// B-matrix 32x16 bf16: lane L, g=L>>4, n=L&15 holds K in the same order as A (8g+0..7, 16+8g+0..7).
union FragU {
    v16bf v;
    uint4 q[2];
};

static __device__ __forceinline__ v16bf ld_frag_a(const __bf16* p /* &A[m*lda + k0 + 8*g] */) {
    FragU u;
    u.q[0] = *(const uint4*)(p);
    u.q[1] = *(const uint4*)(p + 16);
    return u.v;
}
static __device__ __forceinline__ v16bf ld_frag_b(const __bf16* p /* &B[n*ldb + k0 + 8*g] */) {
    FragU u;
    u.q[0] = *(const uint4*)(p);
    u.q[1] = *(const uint4*)(p + 16);
    return u.v;
}
static __device__ __forceinline__ v16bf zero_frag() {
    v16bf v;
#pragma unroll
    for (int i = 0; i < 16; ++i) v[i] = (__bf16)0.0f;
    return v;
}
static __device__ __forceinline__ v8f zero_acc() {
    v8f a;
#pragma unroll
    for (int i = 0; i < 8; ++i) a[i] = 0.0f;
    return a;
}
static __device__ __forceinline__ v8f wmma_bf16(v16bf a, v16bf b, v8f c) {
    return __builtin_amdgcn_wmma_f32_16x16x32_bf16(false, a, false, b, (short)0, c, false, false);
}

// gfx1250 has a hardware transcendental TANH; use it when the builtin exists.
static __device__ __forceinline__ float fast_tanh(float x) {
#if __has_builtin(__builtin_amdgcn_tanhf)
    return __builtin_amdgcn_tanhf(x);
#elif __has_builtin(__builtin_amdgcn_tanh_f32)
    return __builtin_amdgcn_tanh_f32(x);
#else
    float e = __expf(2.0f * x);
    return (e - 1.0f) / (e + 1.0f);
#endif
}
static __device__ __forceinline__ float sigf(float x) { return 1.0f / (1.0f + __expf(-x)); }

// ---------------- precompute / conversion kernels ----------------

// features NCHW f32 -> NHWC bf16 (contiguous C for WMMA A-frags + glimpse)
__global__ void cvt_feat_kernel(const float* __restrict__ src, __bf16* __restrict__ dst) {
    int idx = blockIdx.x * blockDim.x + threadIdx.x;
    if (idx >= BQ * CQ * HWQ) return;
    int x = idx & 31;
    int y = (idx >> 5) & 7;
    int c = (idx >> 8) & 511;
    int b = idx >> 17;
    dst[((size_t)b * HWQ + y * WQ + x) * CQ + c] = (__bf16)src[idx];
}

// conv weights (A,C,3,3) f32 -> [tap][a][c] bf16 (contiguous c = K for B-frags)
__global__ void cvt_convw_kernel(const float* __restrict__ src, __bf16* __restrict__ dst) {
    int idx = blockIdx.x * blockDim.x + threadIdx.x;
    if (idx >= AQ * CQ * 9) return;
    int t = idx % 9;
    int rest = idx / 9;
    int c = rest & 511;
    int a = rest >> 9;
    dst[((size_t)t * AQ + a) * CQ + c] = (__bf16)src[idx];
}

__global__ void cvt_copy_kernel(const float* __restrict__ src, __bf16* __restrict__ dst, int n) {
    int idx = blockIdx.x * blockDim.x + threadIdx.x;
    if (idx < n) dst[idx] = (__bf16)src[idx];
}

// out_W (111,1024) -> (112,1024) bf16, last row zero
__global__ void cvt_outw_kernel(const float* __restrict__ src, __bf16* __restrict__ dst) {
    int idx = blockIdx.x * blockDim.x + threadIdx.x;
    if (idx >= 112 * 1024) return;
    int n = idx >> 10;
    int k = idx & 1023;
    dst[idx] = (n < V1Q) ? (__bf16)src[n * 1024 + k] : (__bf16)0.0f;
}

__global__ void bias_sum_kernel(const float* __restrict__ bih, const float* __restrict__ bhh,
                                float* __restrict__ bs) {
    int idx = blockIdx.x * blockDim.x + threadIdx.x;
    if (idx < NGATE) bs[idx] = bih[idx] + bhh[idx];
}

// prev_symbols: ps[t][b][r] bf16, t=0 holistic, t=1 embed_b, t>=2 embedding matmul
__global__ void ps_kernel(const float* __restrict__ holistic, const float* __restrict__ embed_b,
                          const float* __restrict__ embed_W, const float* __restrict__ embed_tgt_W,
                          const int* __restrict__ gt, __bf16* __restrict__ ps) {
    int idx = blockIdx.x * blockDim.x + threadIdx.x;
    if (idx >= 31 * BQ * RQ) return;
    int r = idx & 511;
    int b = (idx >> 9) & 127;
    int t = idx >> 16;  // 128*512 = 2^16
    float v;
    if (t == 0) {
        v = holistic[(size_t)b * RQ + r];
    } else if (t == 1) {
        v = embed_b[r];
    } else {
        int sym = gt[b * (TQ + 1) + (t - 2)];
        const float* e = embed_tgt_W + (size_t)sym * V1Q;
        const float* w = embed_W + (size_t)r * V1Q;
        float acc = embed_b[r];
        for (int k = 0; k < V1Q; ++k) acc += e[k] * w[k];
        v = acc;
    }
    ps[idx] = (__bf16)v;
}

// ---------------- implicit-GEMM 3x3 conv via WMMA + async LDS staging ----------------
// Block: one image row (32 pixels = 32 M-rows) x all 512 output channels.
// 8 waves, each computing a 32x64 tile (8 accumulators).
// A-data: 3-row NHWC slab (y-1,y,y+1) staged into LDS with global_load_async_to_lds_b128,
// reused by all 9 taps and all 8 waves.  B-data (weights) streams from L2.
#define SLAB_ROW (WQ * CQ)       // 16384 bf16 per image row
#define SLAB_ELEMS (3 * SLAB_ROW)

__global__ __launch_bounds__(256) void conv_gemm_kernel(const __bf16* __restrict__ featT,
                                                        const __bf16* __restrict__ convW,
                                                        const float* __restrict__ cbias,
                                                        __bf16* __restrict__ fproj) {
    __shared__ __attribute__((aligned(16))) __bf16 sA[SLAB_ELEMS];  // 96 KB (320 KB WGP LDS)
    const int tid = threadIdx.x;
    const int lane = tid & 31;
    const int wv = tid >> 5;            // wave in block: 0..7
    const int m0 = blockIdx.x * 32;     // first of 32 output pixels = one image row
    const int b = m0 >> 8;
    const int y = (m0 & 255) >> 5;

    // ---- stage slab rows via async global->LDS copies (ASYNCcnt path) ----
#pragma unroll
    for (int r = 0; r < 3; ++r) {
        const int yy = y + r - 1;
        if ((unsigned)yy < HQ) {
            const __bf16* src = featT + ((size_t)b * HWQ + yy * WQ) * CQ;
            const unsigned ldsBase = (unsigned)(size_t)(&sA[r * SLAB_ROW]);
#pragma unroll
            for (int i = 0; i < (SLAB_ROW / 8) / 256; ++i) {  // 8 chunks of 16B per thread
                const int chunk = i * 256 + tid;
                unsigned ldsOff = ldsBase + chunk * 16;
                unsigned long long ga = (unsigned long long)(size_t)(src + chunk * 8);
                asm volatile("global_load_async_to_lds_b128 %0, %1, off"
                             :: "v"(ldsOff), "v"(ga) : "memory");
            }
        } else {
            uint4 z = make_uint4(0u, 0u, 0u, 0u);
#pragma unroll
            for (int i = 0; i < (SLAB_ROW / 8) / 256; ++i) {
                const int chunk = i * 256 + tid;
                *(uint4*)(&sA[r * SLAB_ROW + chunk * 8]) = z;
            }
        }
    }
    asm volatile("s_wait_asynccnt 0x0" ::: "memory");
    __syncthreads();

    // ---- compute: 9 taps x (K=512) with 32x64 per-wave tile ----
    const int g = lane >> 4, ln = lane & 15;
    const int n0 = wv * 64;
    v8f acc[8];
#pragma unroll
    for (int i = 0; i < 8; ++i) acc[i] = zero_acc();

#pragma unroll
    for (int t = 0; t < 9; ++t) {
        const int sr = t / 3;           // slab row (== dy index 0..2)
        const int dx = t % 3 - 1;
        const int x0 = ln + dx;         // x of A rows m0+ln
        const int x1 = 16 + ln + dx;    // x of A rows m0+16+ln
        const bool ok0 = (unsigned)x0 < WQ;
        const bool ok1 = (unsigned)x1 < WQ;
        const __bf16* a0p = &sA[(sr * WQ + x0) * CQ + 8 * g];
        const __bf16* a1p = &sA[(sr * WQ + x1) * CQ + 8 * g];
        const __bf16* bBase = convW + (size_t)t * AQ * CQ;
        for (int k0 = 0; k0 < CQ; k0 += 32) {
            v16bf a0 = ok0 ? ld_frag_a(a0p + k0) : zero_frag();
            v16bf a1 = ok1 ? ld_frag_a(a1p + k0) : zero_frag();
#pragma unroll
            for (int j = 0; j < 4; ++j) {
                v16bf bb = ld_frag_b(bBase + (size_t)(n0 + 16 * j + ln) * CQ + k0 + 8 * g);
                acc[j]     = wmma_bf16(a0, bb, acc[j]);
                acc[4 + j] = wmma_bf16(a1, bb, acc[4 + j]);
            }
        }
    }

#pragma unroll
    for (int j = 0; j < 4; ++j) {
        const int n = n0 + 16 * j + ln;
        const float bias = cbias[n];
#pragma unroll
        for (int e = 0; e < 8; ++e) {
            const int mm = m0 + 8 * g + e;
            fproj[(size_t)mm * AQ + n]        = (__bf16)(acc[j][e] + bias);
            fproj[(size_t)(mm + 16) * AQ + n] = (__bf16)(acc[4 + j][e] + bias);
        }
    }
}

// ---------------- LSTM gates dual-GEMM: gates = X@Wih^T + H@Whh^T + (b_ih+b_hh) ----------------
// M=128, N=2048, K=512.  Wih/Whh stored [n][k] bf16 (row-major = B^T layout).
__global__ __launch_bounds__(256) void gemm_gates_kernel(const __bf16* __restrict__ X, int ldx,
                                                         const __bf16* __restrict__ Wih,
                                                         const __bf16* __restrict__ Hprev, int ldh,
                                                         const __bf16* __restrict__ Whh,
                                                         const float* __restrict__ biasSum,
                                                         float* __restrict__ gates) {
    const int lane = threadIdx.x & 31;
    const int wave = blockIdx.x * 8 + (threadIdx.x >> 5);  // 256 waves
    const int mt = wave >> 5;   // 0..7
    const int ng = wave & 31;   // 0..31
    const int m0 = mt * 16, n0 = ng * 64;
    const int g = lane >> 4, ln = lane & 15;

    v8f acc[4] = {zero_acc(), zero_acc(), zero_acc(), zero_acc()};
    const __bf16* aRow1 = X + (size_t)(m0 + ln) * ldx + 8 * g;
    const __bf16* aRow2 = Hprev + (size_t)(m0 + ln) * ldh + 8 * g;
    for (int k0 = 0; k0 < RQ; k0 += 32) {
        v16bf a1 = ld_frag_a(aRow1 + k0);
#pragma unroll
        for (int j = 0; j < 4; ++j)
            acc[j] = wmma_bf16(a1, ld_frag_b(Wih + (size_t)(n0 + 16 * j + ln) * RQ + k0 + 8 * g),
                               acc[j]);
        v16bf a2 = ld_frag_a(aRow2 + k0);
#pragma unroll
        for (int j = 0; j < 4; ++j)
            acc[j] = wmma_bf16(a2, ld_frag_b(Whh + (size_t)(n0 + 16 * j + ln) * RQ + k0 + 8 * g),
                               acc[j]);
    }
#pragma unroll
    for (int j = 0; j < 4; ++j) {
        int n = n0 + 16 * j + ln;
        float bias = biasSum[n];
#pragma unroll
        for (int e = 0; e < 8; ++e) {
            int m = m0 + 8 * g + e;
            gates[(size_t)m * NGATE + n] = acc[j][e] + bias;
        }
    }
}

// ---------------- LSTM pointwise: c' = sig(f)*c + sig(i)*tanh(g); h' = sig(o)*tanh(c') ------
__global__ void lstm_pointwise_kernel(const float* __restrict__ gates, float* __restrict__ cState,
                                      __bf16* __restrict__ hDst, int hStride) {
    int idx = blockIdx.x * blockDim.x + threadIdx.x;
    if (idx >= BQ * RQ) return;
    int b = idx >> 9, r = idx & 511;
    const float* gr = gates + (size_t)b * NGATE;
    float i_ = sigf(gr[r]);
    float f_ = sigf(gr[512 + r]);
    float g_ = fast_tanh(gr[1024 + r]);
    float o_ = sigf(gr[1536 + r]);
    float c = f_ * cState[idx] + i_ * g_;
    cState[idx] = c;
    hDst[(size_t)b * hStride + r] = (__bf16)(o_ * fast_tanh(c));
}

// ---------------- sp = h @ state_conv_W^T : M=128, N=512, K=512 ----------------
__global__ __launch_bounds__(256) void gemm_sp_kernel(const __bf16* __restrict__ A, int lda,
                                                      const __bf16* __restrict__ Bm,
                                                      float* __restrict__ outp) {
    const int lane = threadIdx.x & 31;
    const int wave = blockIdx.x * 8 + (threadIdx.x >> 5);  // 64 waves
    const int mt = wave >> 3;  // 0..7
    const int ng = wave & 7;   // 0..7
    const int m0 = mt * 16, n0 = ng * 64;
    const int g = lane >> 4, ln = lane & 15;

    v8f acc[4] = {zero_acc(), zero_acc(), zero_acc(), zero_acc()};
    const __bf16* aRow = A + (size_t)(m0 + ln) * lda + 8 * g;
    for (int k0 = 0; k0 < RQ; k0 += 32) {
        v16bf a = ld_frag_a(aRow + k0);
#pragma unroll
        for (int j = 0; j < 4; ++j)
            acc[j] = wmma_bf16(a, ld_frag_b(Bm + (size_t)(n0 + 16 * j + ln) * RQ + k0 + 8 * g),
                               acc[j]);
    }
#pragma unroll
    for (int j = 0; j < 4; ++j) {
        int n = n0 + 16 * j + ln;
#pragma unroll
        for (int e = 0; e < 8; ++e) {
            int m = m0 + 8 * g + e;
            outp[(size_t)m * AQ + n] = acc[j][e];
        }
    }
}

// ---------------- attention: score -> softmax -> glimpse (per batch element) ----------------
__global__ __launch_bounds__(256) void attn_kernel(const __bf16* __restrict__ fproj,
                                                   const __bf16* __restrict__ featT,
                                                   const float* __restrict__ sp,
                                                   const float* __restrict__ attnW,
                                                   __bf16* __restrict__ catBuf) {
    __shared__ float sSp[AQ];
    __shared__ float sWA[AQ];
    __shared__ float sW[HWQ];
    __shared__ float sRed[HWQ];
    const int b = blockIdx.x;
    const int tid = threadIdx.x;

    sSp[tid]       = sp[(size_t)b * AQ + tid];
    sSp[tid + 256] = sp[(size_t)b * AQ + tid + 256];
    sWA[tid]       = attnW[tid];
    sWA[tid + 256] = attnW[tid + 256];
    __syncthreads();

    // score[p] = sum_a attnW[a] * tanh(fproj[b,p,a] + sp[b,a])   (b128-vectorized)
    const uint4* fp4 = (const uint4*)(fproj + ((size_t)b * HWQ + tid) * AQ);
    float score = 0.0f;
    for (int a4 = 0; a4 < AQ / 8; ++a4) {
        union { uint4 q; __bf16 h[8]; } u;
        u.q = fp4[a4];
        const int base = a4 * 8;
#pragma unroll
        for (int i = 0; i < 8; ++i)
            score += sWA[base + i] * fast_tanh((float)u.h[i] + sSp[base + i]);
    }

    // softmax over the 256 spatial positions
    sRed[tid] = score;
    __syncthreads();
    for (int s = 128; s > 0; s >>= 1) {
        if (tid < s) sRed[tid] = fmaxf(sRed[tid], sRed[tid + s]);
        __syncthreads();
    }
    float mx = sRed[0];
    __syncthreads();
    float e = __expf(score - mx);
    sRed[tid] = e;
    __syncthreads();
    for (int s = 128; s > 0; s >>= 1) {
        if (tid < s) sRed[tid] += sRed[tid + s];
        __syncthreads();
    }
    float inv = 1.0f / sRed[0];
    sW[tid] = e * inv;
    __syncthreads();

    // glimpse[c] = sum_p w[p] * featT[b,p,c]  (each thread: 2 channels, b32 packed)
    const int c0 = tid * 2;
    const unsigned* fb = (const unsigned*)(featT + (size_t)b * HWQ * CQ + c0);
    float g0 = 0.0f, g1 = 0.0f;
    for (int p = 0; p < HWQ; ++p) {
        union { unsigned u; __bf16 h[2]; } w2;
        w2.u = fb[(size_t)p * (CQ / 2)];
        float wp = sW[p];
        g0 += wp * (float)w2.h[0];
        g1 += wp * (float)w2.h[1];
    }
    union { unsigned u; __bf16 h[2]; } o;
    o.h[0] = (__bf16)g0;
    o.h[1] = (__bf16)g1;
    ((unsigned*)(catBuf + (size_t)b * 1024 + 512))[tid] = o.u;
}

// ---------------- logits = [h|glimpse] @ out_W^T + out_b : M=128, N=111(pad 112), K=1024 ----
__global__ __launch_bounds__(256) void gemm_logits_kernel(const __bf16* __restrict__ catBuf,
                                                          const __bf16* __restrict__ outWb,
                                                          const float* __restrict__ outb,
                                                          float* __restrict__ dout, int tstep) {
    const int lane = threadIdx.x & 31;
    const int wave = blockIdx.x * 8 + (threadIdx.x >> 5);  // 16 waves
    const int mt = wave >> 1;       // 0..7
    const int half = wave & 1;      // 0..1
    const int m0 = mt * 16, n0 = half * 64;
    const int nt = half ? 3 : 4;    // 7 n-tiles total
    const int g = lane >> 4, ln = lane & 15;

    v8f acc[4] = {zero_acc(), zero_acc(), zero_acc(), zero_acc()};
    const __bf16* aRow = catBuf + (size_t)(m0 + ln) * 1024 + 8 * g;
    for (int k0 = 0; k0 < 1024; k0 += 32) {
        v16bf a = ld_frag_a(aRow + k0);
        for (int j = 0; j < nt; ++j)
            acc[j] = wmma_bf16(a, ld_frag_b(outWb + (size_t)(n0 + 16 * j + ln) * 1024 + k0 + 8 * g),
                               acc[j]);
    }
    for (int j = 0; j < nt; ++j) {
        int n = n0 + 16 * j + ln;
        if (n >= V1Q) continue;
        float bias = outb[n];
#pragma unroll
        for (int e = 0; e < 8; ++e) {
            int m = m0 + 8 * g + e;
            dout[((size_t)m * TQ + (tstep - 1)) * V1Q + n] = acc[j][e] + bias;
        }
    }
}

// ---------------- host orchestration ----------------
extern "C" void kernel_launch(void* const* d_in, const int* in_sizes, int n_in,
                              void* d_out, int out_size, void* d_ws, size_t ws_size,
                              hipStream_t stream) {
    (void)in_sizes; (void)n_in; (void)out_size; (void)ws_size;
    const float* features     = (const float*)d_in[0];
    const float* holistic     = (const float*)d_in[1];
    const int*   gt           = (const int*)d_in[2];
    const float* embed_W      = (const float*)d_in[3];
    const float* embed_b      = (const float*)d_in[4];
    const float* embed_tgt_W  = (const float*)d_in[5];
    const float* feat_conv_W  = (const float*)d_in[6];
    const float* feat_conv_b  = (const float*)d_in[7];
    const float* state_conv_W = (const float*)d_in[8];
    const float* attn_proj_W  = (const float*)d_in[9];
    const float* W_ih         = (const float*)d_in[10];
    const float* W_hh         = (const float*)d_in[11];
    const float* b_ih         = (const float*)d_in[12];
    const float* b_hh         = (const float*)d_in[13];
    const float* out_W        = (const float*)d_in[14];
    const float* out_b        = (const float*)d_in[15];
    float* dout = (float*)d_out;

    // carve workspace (~82 MB)
    uintptr_t base = (uintptr_t)d_ws;
    auto alloc = [&](size_t bytes) -> void* {
        uintptr_t p = (base + 255) & ~(uintptr_t)255;
        base = p + bytes;
        return (void*)p;
    };
    __bf16* featT   = (__bf16*)alloc((size_t)BQ * HWQ * CQ * 2);     // NHWC bf16 features
    __bf16* fproj   = (__bf16*)alloc((size_t)BQ * HWQ * AQ * 2);     // conv output [b][p][a]
    __bf16* convWb  = (__bf16*)alloc((size_t)9 * AQ * CQ * 2);
    __bf16* WihB    = (__bf16*)alloc((size_t)NGATE * RQ * 2);
    __bf16* WhhB    = (__bf16*)alloc((size_t)NGATE * RQ * 2);
    __bf16* stWB    = (__bf16*)alloc((size_t)AQ * RQ * 2);
    __bf16* outWb   = (__bf16*)alloc((size_t)112 * 1024 * 2);
    float*  biasSum = (float*)alloc((size_t)NGATE * 4);
    __bf16* ps      = (__bf16*)alloc((size_t)31 * BQ * RQ * 2);
    float*  gatesB  = (float*)alloc((size_t)BQ * NGATE * 4);
    __bf16* hiB     = (__bf16*)alloc((size_t)BQ * RQ * 2);           // inner-cell hidden (bf16)
    float*  ci      = (float*)alloc((size_t)BQ * RQ * 4);            // inner-cell c state
    float*  cc      = (float*)alloc((size_t)BQ * RQ * 4);            // outer-cell c state
    __bf16* catB    = (__bf16*)alloc((size_t)BQ * 1024 * 2);         // [h | glimpse] bf16
    float*  spB     = (float*)alloc((size_t)BQ * AQ * 4);

    const int thr = 256;

    // one-time conversions / precompute
    cvt_feat_kernel<<<(BQ * CQ * HWQ) / thr, thr, 0, stream>>>(features, featT);
    cvt_convw_kernel<<<(AQ * CQ * 9 + thr - 1) / thr, thr, 0, stream>>>(feat_conv_W, convWb);
    cvt_copy_kernel<<<(NGATE * RQ) / thr, thr, 0, stream>>>(W_ih, WihB, NGATE * RQ);
    cvt_copy_kernel<<<(NGATE * RQ) / thr, thr, 0, stream>>>(W_hh, WhhB, NGATE * RQ);
    cvt_copy_kernel<<<(AQ * RQ) / thr, thr, 0, stream>>>(state_conv_W, stWB, AQ * RQ);
    cvt_outw_kernel<<<(112 * 1024) / thr, thr, 0, stream>>>(out_W, outWb);
    bias_sum_kernel<<<NGATE / thr, thr, 0, stream>>>(b_ih, b_hh, biasSum);
    ps_kernel<<<(31 * BQ * RQ + thr - 1) / thr, thr, 0, stream>>>(holistic, embed_b, embed_W,
                                                                  embed_tgt_W, gt, ps);
    // heavy WMMA conv: 1024 blocks, 32x64 per-wave tiles, async-LDS staged A
    conv_gemm_kernel<<<1024, 256, 0, stream>>>(featT, convWb, feat_conv_b, fproj);

    // zero LSTM state (graph-capturable memsets)
    hipMemsetAsync(hiB, 0, (size_t)BQ * RQ * 2, stream);
    hipMemsetAsync(ci, 0, (size_t)BQ * RQ * 4, stream);
    hipMemsetAsync(cc, 0, (size_t)BQ * RQ * 4, stream);
    hipMemsetAsync(catB, 0, (size_t)BQ * 1024 * 2, stream);

    // recurrent scan: 31 steps; logits stored for steps 1..30
    for (int t = 0; t <= TQ; ++t) {
        // cell 1: gates = ps_t @ Wih^T + hi @ Whh^T + bias
        gemm_gates_kernel<<<32, 256, 0, stream>>>(ps + (size_t)t * BQ * RQ, RQ, WihB,
                                                  hiB, RQ, WhhB, biasSum, gatesB);
        lstm_pointwise_kernel<<<(BQ * RQ) / thr, thr, 0, stream>>>(gatesB, ci, hiB, RQ);
        // cell 2: gates = hi @ Wih^T + h @ Whh^T + bias   (h lives in catB[:,0:512], ld=1024)
        gemm_gates_kernel<<<32, 256, 0, stream>>>(hiB, RQ, WihB, catB, 1024, WhhB, biasSum, gatesB);
        lstm_pointwise_kernel<<<(BQ * RQ) / thr, thr, 0, stream>>>(gatesB, cc, catB, 1024);
        if (t == 0) continue;  // step-0 logits are discarded by the reference
        // attention + readout
        gemm_sp_kernel<<<8, 256, 0, stream>>>(catB, 1024, stWB, spB);
        attn_kernel<<<BQ, 256, 0, stream>>>(fproj, featT, spB, attn_proj_W, catB);
        gemm_logits_kernel<<<2, 256, 0, stream>>>(catB, outWb, out_b, dout, t);
    }
}
